// GAT_40080634806960
// MI455X (gfx1250) — matrix operand order repaired and checked by reading, not compiled, not run on hardware
//
#include <hip/hip_runtime.h>
#include <hip/hip_bf16.h>

#define N_NODES 50000
#define N_EDGES 800000
#define D_MODEL 96
#define HEADS   3
#define HID     32

typedef _Float16 v16h __attribute__((ext_vector_type(16)));
typedef float    v8f  __attribute__((ext_vector_type(8)));

// ---------------------------------------------------------------- helpers
__device__ __forceinline__ void atomicMaxF(float* addr, float v) {
    // int/uint trick: valid for mixed-sign floats when initialized to -inf
    if (v >= 0.f) atomicMax((int*)addr, __float_as_int(v));
    else          atomicMin((unsigned int*)addr, __float_as_uint(v));
}

// A-fragment (16x32 f16, ISA 7.12.2) flat index for element (row, kc) of
// k-block kb in node-tile `tile`; storage is [tile][kb][lane][16] halfs.
// lane<16 holds K {0..7,16..23}; lane>=16 holds K {8..15,24..31}.
__device__ __forceinline__ size_t fragIdxA(int tile, int kb, int KB, int row, int kc) {
    int group  = kc >> 3;                       // 0..3
    int half16 = group & 1;                     // which lane half
    int slot   = ((group >> 1) << 3) + (kc & 7);
    return ((((size_t)tile * KB + kb) * 32) + half16 * 16 + row) * 16 + slot;
}

// ---------------------------------------------------------------- input MLP
// h0 = tanh(x @ W1 + b1) written directly into A-fragment layout (KB=1,
// K padded to 32 with zeros).
__global__ void input_mlp(const float* __restrict__ x, const float* __restrict__ W1,
                          const float* __restrict__ b1, _Float16* __restrict__ H0) {
    int i = blockIdx.x * blockDim.x + threadIdx.x;
    if (i >= N_NODES) return;
    float xv[16];
#pragma unroll
    for (int j = 0; j < 16; ++j) xv[j] = x[(size_t)i * 16 + j];
    const int tile = i >> 4, row = i & 15;
#pragma unroll
    for (int o = 0; o < 8; ++o) {
        float s = b1[o];
#pragma unroll
        for (int j = 0; j < 16; ++j) s += xv[j] * W1[j * 8 + o];
        H0[fragIdxA(tile, 0, 1, row, o)] = (_Float16)tanhf(s);
    }
#pragma unroll
    for (int kc = 8; kc < 32; ++kc) H0[fragIdxA(tile, 0, 1, row, kc)] = (_Float16)0.f;
}

// --------------------------------------------- weight pack -> B-fragment f16
// dst layout: [mat(4)][ntile(6)][kb(KB)][lane(32)][slot(16)] halfs.
// col n = nt*16+(lane&15); K = kb*32 + (slot>>3)*16 + (lane>>4)*8 + (slot&7).
__global__ void pack_w(const float* __restrict__ Wq, const float* __restrict__ Wk,
                       const float* __restrict__ Wv, const float* __restrict__ Ws,
                       _Float16* __restrict__ dst, int KB, int Ksrc) {
    int idx = blockIdx.x * blockDim.x + threadIdx.x;
    const int per_mat = 6 * KB * 512;
    if (idx >= 4 * per_mat) return;
    int mat = idx / per_mat;
    int r   = idx % per_mat;
    int nt  = r / (KB * 512); r %= KB * 512;
    int kb  = r / 512;        r %= 512;
    int l   = r >> 4;
    int slot = r & 15;
    int k = kb * 32 + (slot >> 3) * 16 + (l >> 4) * 8 + (slot & 7);
    int n = nt * 16 + (l & 15);
    const float* W = (mat == 0) ? Wq : (mat == 1) ? Wk : (mat == 2) ? Wv : Ws;
    dst[idx] = (_Float16)((k < Ksrc) ? W[k * D_MODEL + n] : 0.f);
}

// ----------------------------------------------------------- per-layer init
__global__ void init_attn(float* __restrict__ M, float* __restrict__ S,
                          float* __restrict__ A) {
    long idx = (long)blockIdx.x * blockDim.x + threadIdx.x;
    if (idx < (long)N_NODES * D_MODEL) A[idx] = 0.f;
    if (idx < (long)N_NODES * HEADS) { M[idx] = -__builtin_inff(); S[idx] = 0.f; }
}

// ------------------------------------------------------------- WMMA GEMM
// One wave = one 16-row node stripe of one matrix: keeps A fragments in
// registers and sweeps all 6 column tiles. blockIdx: x = node tile, y = mat.
// All four outputs are f16, so the destination pointer is selected ONCE
// (uniform) and the store loop is branch-free.
template <int KPAD>
__global__ void gemm_wmma(const _Float16* __restrict__ Hf,
                          const _Float16* __restrict__ Wf,
                          const float* __restrict__ bq, const float* __restrict__ bk,
                          const float* __restrict__ bv, const float* __restrict__ bs,
                          _Float16* __restrict__ Qo, _Float16* __restrict__ Ko,
                          _Float16* __restrict__ Vo, _Float16* __restrict__ SKo) {
    constexpr int KB = KPAD / 32;
    const int lane   = threadIdx.x;
    const int l15    = lane & 15;
    const int half16 = lane >> 4;
    const int tile   = blockIdx.x;
    const int mat    = blockIdx.y;

    // A fragments for this 16-row stripe (contiguous 32B per lane)
    v16h a[KB];
#pragma unroll
    for (int kb = 0; kb < KB; ++kb)
        a[kb] = *(const v16h*)(Hf + (((size_t)tile * KB + kb) * 32 + lane) * 16);

    const _Float16* Wm   = Wf + (size_t)mat * (6 * KB * 512);
    const float*    bias = (mat == 0) ? bq : (mat == 1) ? bk : (mat == 2) ? bv : bs;
    _Float16*       out  = (mat == 0) ? Qo : (mat == 1) ? Ko : (mat == 2) ? Vo : SKo;
    const int       rbase = tile * 16 + half16 * 8;

#pragma unroll
    for (int nt = 0; nt < 6; ++nt) {
        v8f acc = {};
#pragma unroll
        for (int kb = 0; kb < KB; ++kb) {
            v16h b = *(const v16h*)(Wm + (((size_t)nt * KB + kb) * 32 + lane) * 16);
            acc = __builtin_amdgcn_wmma_f32_16x16x32_f16(
                false, a[kb], false, b, (short)0, acc, false, false);
        }
        const int   col = nt * 16 + l15;
        const float bb  = bias[col];
#pragma unroll
        for (int r = 0; r < 8; ++r)
            out[(size_t)(rbase + r) * D_MODEL + col] = (_Float16)(acc[r] + bb);
    }
}

// ----------------------------------------------- edge pass 1: logits -> max
// wave32 per (edge, head): lane = component d in [0,32)
__global__ void edge_pass1(const _Float16* __restrict__ Q, const _Float16* __restrict__ Km,
                           const int* __restrict__ src, const int* __restrict__ dst,
                           const float* __restrict__ eattr, const float* __restrict__ We,
                           float* __restrict__ M) {
    const int  lane = threadIdx.x;
    const long pair = (long)blockIdx.x * blockDim.y + threadIdx.y;
    if (pair >= (long)N_EDGES * HEADS) return;
    const int e = (int)(pair / HEADS), h = (int)(pair % HEADS);
    const int s = src[e], d = dst[e];
    const int c = h * HID + lane;
    const float e0 = eattr[(size_t)e * 4 + 0], e1 = eattr[(size_t)e * 4 + 1];
    const float e2 = eattr[(size_t)e * 4 + 2], e3 = eattr[(size_t)e * 4 + 3];
    const float ee = e0 * We[c] + e1 * We[96 + c] + e2 * We[192 + c] + e3 * We[288 + c];
    const float qv = (float)Q[(size_t)d * D_MODEL + c];
    const float kv = (float)Km[(size_t)s * D_MODEL + c] + ee;
    float t = qv * kv;
#pragma unroll
    for (int off = 16; off; off >>= 1) t += __shfl_xor(t, off, 32);
    t *= 0.1767766952966369f; // 1/sqrt(32)
    if (lane == 0) atomicMaxF(&M[(size_t)d * HEADS + h], t);
}

// --------------------------------- edge pass 2: p = exp(l-m); scatter p, p*v
__global__ void edge_pass2(const _Float16* __restrict__ Q, const _Float16* __restrict__ Km,
                           const _Float16* __restrict__ Vm,
                           const int* __restrict__ src, const int* __restrict__ dst,
                           const float* __restrict__ eattr, const float* __restrict__ We,
                           const float* __restrict__ M, float* __restrict__ S,
                           float* __restrict__ A) {
    const int  lane = threadIdx.x;
    const long pair = (long)blockIdx.x * blockDim.y + threadIdx.y;
    if (pair >= (long)N_EDGES * HEADS) return;
    const int e = (int)(pair / HEADS), h = (int)(pair % HEADS);
    const int s = src[e], d = dst[e];
    const int c = h * HID + lane;
    const float e0 = eattr[(size_t)e * 4 + 0], e1 = eattr[(size_t)e * 4 + 1];
    const float e2 = eattr[(size_t)e * 4 + 2], e3 = eattr[(size_t)e * 4 + 3];
    const float ee = e0 * We[c] + e1 * We[96 + c] + e2 * We[192 + c] + e3 * We[288 + c];
    const float qv = (float)Q[(size_t)d * D_MODEL + c];
    const float kv = (float)Km[(size_t)s * D_MODEL + c] + ee;
    float t = qv * kv;
#pragma unroll
    for (int off = 16; off; off >>= 1) t += __shfl_xor(t, off, 32);
    t *= 0.1767766952966369f;
    const float p = __expf(t - M[(size_t)d * HEADS + h]);
    if (lane == 0) atomicAdd(&S[(size_t)d * HEADS + h], p);
    const float vv = (float)Vm[(size_t)s * D_MODEL + c] + ee;
    atomicAdd(&A[(size_t)d * D_MODEL + c], p * vv);
}

// ------------------------------------------- finalize: ÷sum, +skip, tanh
// mode 0: write f16 A-fragment layout (KB=3) for the next layer's GEMM
// mode 1: write f32 row-major to d_out
__global__ void finalize_layer(const float* __restrict__ A, const float* __restrict__ S,
                               const _Float16* __restrict__ SK,
                               _Float16* __restrict__ outH,
                               float* __restrict__ outF, int mode) {
    long idx = (long)blockIdx.x * blockDim.x + threadIdx.x;
    if (idx >= (long)N_NODES * D_MODEL) return;
    int n = (int)(idx / D_MODEL), c = (int)(idx % D_MODEL);
    int h = c / HID;
    float a = A[idx] / (S[(size_t)n * HEADS + h] + 1e-16f);
    float o = tanhf(a + (float)SK[idx]);
    if (mode == 0) outH[fragIdxA(n >> 4, c >> 5, 3, n & 15, c & 31)] = (_Float16)o;
    else           outF[idx] = o;
}

extern "C" void kernel_launch(void* const* d_in, const int* in_sizes, int n_in,
                              void* d_out, int out_size, void* d_ws, size_t ws_size,
                              hipStream_t stream) {
    const float* x     = (const float*)d_in[0];
    const int*   ei    = (const int*)d_in[1];
    const float* eattr = (const float*)d_in[2];
    const float* W1    = (const float*)d_in[3];
    const float* b1    = (const float*)d_in[4];
    const float* Wq1 = (const float*)d_in[5],  *bq1 = (const float*)d_in[6];
    const float* Wk1 = (const float*)d_in[7],  *bk1 = (const float*)d_in[8];
    const float* Wv1 = (const float*)d_in[9],  *bv1 = (const float*)d_in[10];
    const float* We1 = (const float*)d_in[11];
    const float* Ws1 = (const float*)d_in[12], *bs1 = (const float*)d_in[13];
    const float* Wq2 = (const float*)d_in[14], *bq2 = (const float*)d_in[15];
    const float* Wk2 = (const float*)d_in[16], *bk2 = (const float*)d_in[17];
    const float* Wv2 = (const float*)d_in[18], *bv2 = (const float*)d_in[19];
    const float* We2 = (const float*)d_in[20];
    const float* Ws2 = (const float*)d_in[21], *bs2 = (const float*)d_in[22];
    const int* srcp = ei;
    const int* dstp = ei + N_EDGES;

    // workspace layout (bytes), all 32B-aligned; total ~81.3 MB
    char* ws = (char*)d_ws;
    _Float16* H0   = (_Float16*)(ws + 0);          // N*32 f16 fragment layout
    _Float16* H1   = (_Float16*)(ws + 3200000);    // N*96 f16 fragment layout
    _Float16* Q16  = (_Float16*)(ws + 12800000);   // N*96 f16 row-major
    _Float16* K16  = (_Float16*)(ws + 22400000);   // N*96 f16 row-major
    _Float16* V16  = (_Float16*)(ws + 32000000);   // N*96 f16 row-major
    _Float16* SK16 = (_Float16*)(ws + 41600000);   // N*96 f16 row-major (skip)
    float*    MBUF = (float*)   (ws + 60800000);   // N*3  f32
    float*    SBUF = (float*)   (ws + 61400000);   // N*3  f32
    float*    ACC  = (float*)   (ws + 62000000);   // N*96 f32
    _Float16* W16  = (_Float16*)(ws + 81200000);   // packed fragment weights

    const int gNode = (N_NODES + 255) / 256;                 // 196
    const int gNC   = (N_NODES * D_MODEL + 255) / 256;       // 18750
    const int gPair = (N_EDGES * HEADS + 7) / 8;             // 300000
    const dim3 gemmGrid(N_NODES / 16, 4);                    // 3125 x 4
    const dim3 edgeBlk(32, 8, 1);

    input_mlp<<<gNode, 256, 0, stream>>>(x, W1, b1, H0);
    pack_w<<<48, 256, 0, stream>>>(Wq1, Wk1, Wv1, Ws1, W16, 1, 8);                 // 12288
    pack_w<<<144, 256, 0, stream>>>(Wq2, Wk2, Wv2, Ws2, W16 + 4 * 6 * 512, 3, 96); // 36864

    // ----- layer 1 (fi = 8, KPAD = 32) -----
    init_attn<<<gNC, 256, 0, stream>>>(MBUF, SBUF, ACC);
    gemm_wmma<32><<<gemmGrid, 32, 0, stream>>>(H0, W16, bq1, bk1, bv1, bs1,
                                               Q16, K16, V16, SK16);
    edge_pass1<<<gPair, edgeBlk, 0, stream>>>(Q16, K16, srcp, dstp, eattr, We1, MBUF);
    edge_pass2<<<gPair, edgeBlk, 0, stream>>>(Q16, K16, V16, srcp, dstp, eattr, We1,
                                              MBUF, SBUF, ACC);
    finalize_layer<<<gNC, 256, 0, stream>>>(ACC, SBUF, SK16, H1, nullptr, 0);

    // ----- layer 2 (fi = 96, KPAD = 96) -----
    init_attn<<<gNC, 256, 0, stream>>>(MBUF, SBUF, ACC);
    gemm_wmma<96><<<gemmGrid, 32, 0, stream>>>(H1, W16 + 4 * 6 * 512, bq2, bk2, bv2, bs2,
                                               Q16, K16, V16, SK16);
    edge_pass1<<<gPair, edgeBlk, 0, stream>>>(Q16, K16, srcp, dstp, eattr, We2, MBUF);
    edge_pass2<<<gPair, edgeBlk, 0, stream>>>(Q16, K16, V16, srcp, dstp, eattr, We2,
                                              MBUF, SBUF, ACC);
    finalize_layer<<<gNC, 256, 0, stream>>>(ACC, SBUF, SK16, nullptr, (float*)d_out, 1);
}
